// NodeMatcher_7121055777017
// MI455X (gfx1250) — matrix operand order repaired and checked by reading, not compiled, not run on hardware
//
#include <hip/hip_runtime.h>

// NodeMatcher for MI455X (gfx1250, wave32).
// bf16 hi/lo split-GEMM (3x v_wmma_f32_16x16x32_bf16 per tile) => fp32-grade
// accuracy at bf16 matrix throughput. Both GEMM kernels stage bf16 operands
// with GLOBAL_LOAD_ASYNC_TO_LDS_B128 (ASYNCcnt) double-buffered pipelines.
// Workspace layout (requires ws_size >= ~169 MB):
//   [0,16MB)      Qh  bf16[8192][1024]
//   [16,32MB)     Ql
//   [32,96MB)     Kh  bf16[32768][1024]
//   [96,160MB)    Kl
//   [160MB,+64KB) mkey u64[8192]  (packed argmax keys)
//   [161MB..169MB) Wqh, Wql, Wkh, Wkl  bf16[1024][1024] each

typedef __attribute__((ext_vector_type(16))) __bf16 v16bf;
typedef __attribute__((ext_vector_type(8)))  float  v8f;
typedef int v4i_ __attribute__((vector_size(16)));   // matches async-LDS builtin param type

#if defined(__has_builtin)
#if __has_builtin(__builtin_amdgcn_global_load_async_to_lds_b128) && \
    __has_builtin(__builtin_amdgcn_s_wait_asynccnt)
#define USE_ASYNC_LDS 1
#endif
#if __has_builtin(__builtin_amdgcn_cvt_pk_bf16_f32)
#define HAVE_CVT_PK_BF16 1
#endif
#endif

constexpr int BATCH = 16;
constexpr int NT    = 512;
constexpr int NS    = 2048;
constexpr int DIM   = 1024;

constexpr int BM = 128;      // block tile M
constexpr int BN = 64;       // block tile N
constexpr int BK = 32;       // k-chunk (bf16 WMMA K)
constexpr int AST = BK + 16; // LDS row stride in u16 (96B: 16B-aligned rows, staggered banks)
constexpr int NK = DIM / BK;
constexpr int NLOAD_S = 6;   // async copies per thread per k-chunk (scores kernel)
constexpr int NLOAD_P = 2;   // async copies per thread per k-chunk (proj B staging)

__device__ __forceinline__ unsigned short f2bf(float x) {
  unsigned u = __float_as_uint(x);
  u += 0x7FFFu + ((u >> 16) & 1u);   // round-to-nearest-even
  return (unsigned short)(u >> 16);
}
__device__ __forceinline__ float bf2f(unsigned short h) {
  return __uint_as_float(((unsigned)h) << 16);
}

// Split two adjacent f32 into packed-bf16 (hi-pair, lo-pair); a -> low halves.
__device__ __forceinline__ void split_pair(float a, float b, unsigned& hp, unsigned& lp) {
#ifdef HAVE_CVT_PK_BF16
  hp = __builtin_bit_cast(unsigned, __builtin_amdgcn_cvt_pk_bf16_f32(a, b));
  float ha = __uint_as_float(hp << 16);
  float hb = __uint_as_float(hp & 0xFFFF0000u);
  lp = __builtin_bit_cast(unsigned, __builtin_amdgcn_cvt_pk_bf16_f32(a - ha, b - hb));
#else
  unsigned short h0 = f2bf(a), h1 = f2bf(b);
  unsigned short l0 = f2bf(a - bf2f(h0)), l1 = f2bf(b - bf2f(h1));
  hp = (unsigned)h0 | ((unsigned)h1 << 16);
  lp = (unsigned)l0 | ((unsigned)l1 << 16);
#endif
}

union Frag { v16bf v; uint4 q[2]; };

__device__ __forceinline__ unsigned long long shflx64(unsigned long long v, int m) {
  unsigned lo = (unsigned)__shfl_xor((int)(unsigned)(v & 0xFFFFFFFFull), m, 32);
  unsigned hi = (unsigned)__shfl_xor((int)(unsigned)(v >> 32), m, 32);
  return (((unsigned long long)hi) << 32) | (unsigned long long)lo;
}

// Pack (score, col) so that u64 max == (max score, then lowest col) — matches
// jnp.argmax first-occurrence tie-breaking.
__device__ __forceinline__ unsigned long long pack_key(float f, int s) {
  unsigned u = __float_as_uint(f);
  u = (u & 0x80000000u) ? ~u : (u | 0x80000000u);
  return (((unsigned long long)u) << 32) | (unsigned)(NS - 1 - s);
}

// 16B global->LDS copy: async (ASYNCcnt) when the gfx1250 builtin exists.
__device__ __forceinline__ void cp16_g2l(const unsigned short* g, unsigned short* l) {
#ifdef USE_ASYNC_LDS
  __builtin_amdgcn_global_load_async_to_lds_b128((v4i_*)g, (v4i_*)l, 0, 0);
#else
  *(uint4*)l = *(const uint4*)g;
#endif
}

template <int N> __device__ __forceinline__ void wait_async() {
#ifdef USE_ASYNC_LDS
  __builtin_amdgcn_s_wait_asynccnt(N);
#endif
}

__device__ __forceinline__ void load16f(float* v, const float* p) {
  const float4* s = (const float4*)p;
#pragma unroll
  for (int j = 0; j < 4; ++j) {
    float4 f = s[j];
    v[4*j+0] = f.x; v[4*j+1] = f.y; v[4*j+2] = f.z; v[4*j+3] = f.w;
  }
}
__device__ __forceinline__ void load8f(float* v, const float* p) {
  const float4* s = (const float4*)p;
#pragma unroll
  for (int j = 0; j < 2; ++j) {
    float4 f = s[j];
    v[4*j+0] = f.x; v[4*j+1] = f.y; v[4*j+2] = f.z; v[4*j+3] = f.w;
  }
}
// Convert COUNT f32 to packed (hi,lo) bf16 pairs and store to LDS.
template <int COUNT>
__device__ __forceinline__ void split_store(const float* v, unsigned short* ph, unsigned short* pl) {
  unsigned* dh = (unsigned*)ph;
  unsigned* dl = (unsigned*)pl;
#pragma unroll
  for (int j = 0; j < COUNT / 2; ++j) split_pair(v[2*j], v[2*j+1], dh[j], dl[j]);
}

// ---------------------------------------------------------------------------
// One-shot elementwise hi/lo split of a weight matrix (1024x1024).
// grid = DIM*DIM/(256*8) blocks; 8 f32 per thread, b128 in / 2x b128 out.
// ---------------------------------------------------------------------------
__global__ __launch_bounds__(256)
void wsplit_kernel(const float* __restrict__ W,
                   unsigned short* __restrict__ Wh, unsigned short* __restrict__ Wl) {
  size_t base = ((size_t)blockIdx.x * 256 + threadIdx.x) * 8;
  float v[8];
  load8f(v, W + base);
  unsigned ph[4], pl[4];
#pragma unroll
  for (int j = 0; j < 4; ++j) split_pair(v[2*j], v[2*j+1], ph[j], pl[j]);
  *(uint4*)(Wh + base) = make_uint4(ph[0], ph[1], ph[2], ph[3]);
  *(uint4*)(Wl + base) = make_uint4(pl[0], pl[1], pl[2], pl[3]);
}

// ---------------------------------------------------------------------------
// P[M,N] = X[M,D] @ W[N,D]^T + b[N], emitted as split bf16 (Ph, Pl).
// grid = (DIM/BN, M/BM), block = 256 (8 waves: 4 along M x 2 along N).
// A (X): register-pipelined f32 loads + on-the-fly split (single LDS buffer).
// B (W): pre-split bf16, async global->LDS b128 copies, double-buffered.
// ---------------------------------------------------------------------------
__global__ __launch_bounds__(256)
void proj_split_kernel(const float* __restrict__ X,
                       const unsigned short* __restrict__ Wh, const unsigned short* __restrict__ Wl,
                       const float* __restrict__ bias,
                       unsigned short* __restrict__ Ph, unsigned short* __restrict__ Pl) {
  __shared__ unsigned short sAh[BM * AST], sAl[BM * AST];
  __shared__ unsigned short sBh[2][BN * AST], sBl[2][BN * AST];

  const int tid   = threadIdx.x;
  const int lane  = tid & 31;
  const int wid   = tid >> 5;
  const int waveM = wid & 3;
  const int waveN = wid >> 2;
  const size_t m0 = (size_t)blockIdx.y * BM;
  const int    n0 = blockIdx.x * BN;

  const int arow = tid >> 1, ahalf = tid & 1;   // A staging: 2 threads/row, 16 f32 each
  const int brow = tid >> 2, bq = tid & 3;      // B staging: 4 threads/row, 8 bf16 each
  const float* aSrc = X + (m0 + arow) * DIM + ahalf * 16;
  const unsigned short* gWh = Wh + (size_t)(n0 + brow) * DIM + bq * 8;
  const unsigned short* gWl = Wl + (size_t)(n0 + brow) * DIM + bq * 8;
  const int bOff = brow * AST + bq * 8;

  v8f acc[2][2] = {};

  const int r16    = lane & 15;
  const int khalfA = (lane >> 4) * 16;  // A: lanes 0-15 -> K 0..15, 16-31 -> K 16..31
  const int khalfB = (lane >> 4) * 16;  // B: lanes 0-15 -> K 0..15, 16-31 -> K 16..31

  float va[16];
  load16f(va, aSrc);
  cp16_g2l(gWh, &sBh[0][bOff]);        // stage B chunk 0
  cp16_g2l(gWl, &sBl[0][bOff]);

  for (int kc = 0; kc < NK; ++kc) {
    const int cur = kc & 1;
    __syncthreads();  // previous chunk's compute done; LDS free to overwrite
    split_store<16>(va, &sAh[arow * AST + ahalf * 16], &sAl[arow * AST + ahalf * 16]);
    if (kc + 1 < NK) {            // next chunk's loads fly during this chunk's WMMAs
      const int k0 = (kc + 1) * BK;
      load16f(va, aSrc + k0);
      cp16_g2l(gWh + k0, &sBh[cur ^ 1][bOff]);
      cp16_g2l(gWl + k0, &sBl[cur ^ 1][bOff]);
      wait_async<NLOAD_P>();      // chunk kc's B copies retired (in-order)
      if (kc + 2 < NK) __builtin_prefetch(aSrc + (kc + 2) * BK, 0, 0);
    } else {
      wait_async<0>();
    }
    __syncthreads();

    Frag ah[2], al[2], bh[2], bl[2];
#pragma unroll
    for (int mt = 0; mt < 2; ++mt) {
      const unsigned short* bh_ = &sAh[(waveM * 32 + mt * 16 + r16) * AST];
      const unsigned short* bl_ = &sAl[(waveM * 32 + mt * 16 + r16) * AST];
      ah[mt].q[0] = *(const uint4*)(bh_ + khalfA);
      ah[mt].q[1] = *(const uint4*)(bh_ + 8 + khalfA);
      al[mt].q[0] = *(const uint4*)(bl_ + khalfA);
      al[mt].q[1] = *(const uint4*)(bl_ + 8 + khalfA);
    }
#pragma unroll
    for (int nt = 0; nt < 2; ++nt) {
      const unsigned short* bh_ = &sBh[cur][(waveN * 32 + nt * 16 + r16) * AST + khalfB];
      const unsigned short* bl_ = &sBl[cur][(waveN * 32 + nt * 16 + r16) * AST + khalfB];
      bh[nt].q[0] = *(const uint4*)(bh_);
      bh[nt].q[1] = *(const uint4*)(bh_ + 8);
      bl[nt].q[0] = *(const uint4*)(bl_);
      bl[nt].q[1] = *(const uint4*)(bl_ + 8);
    }
#pragma unroll
    for (int mt = 0; mt < 2; ++mt)
#pragma unroll
      for (int nt = 0; nt < 2; ++nt) {
        acc[mt][nt] = __builtin_amdgcn_wmma_f32_16x16x32_bf16(
            false, ah[mt].v, false, bh[nt].v, (short)0, acc[mt][nt], false, false);
        acc[mt][nt] = __builtin_amdgcn_wmma_f32_16x16x32_bf16(
            false, ah[mt].v, false, bl[nt].v, (short)0, acc[mt][nt], false, false);
        acc[mt][nt] = __builtin_amdgcn_wmma_f32_16x16x32_bf16(
            false, al[mt].v, false, bh[nt].v, (short)0, acc[mt][nt], false, false);
      }
  }

  // epilogue: +bias, split to hi/lo bf16, store
#pragma unroll
  for (int mt = 0; mt < 2; ++mt)
#pragma unroll
    for (int nt = 0; nt < 2; ++nt) {
      int n = n0 + waveN * 32 + nt * 16 + (lane & 15);
      float bb = bias[n];
      size_t mbase = m0 + waveM * 32 + mt * 16 + ((lane >> 4) * 8);
#pragma unroll
      for (int v = 0; v < 8; ++v) {
        float s = acc[mt][nt][v] + bb;
        size_t idx = (mbase + v) * DIM + n;
        unsigned short h = f2bf(s);
        Ph[idx] = h;
        Pl[idx] = f2bf(s - bf2f(h));
      }
    }
}

// ---------------------------------------------------------------------------
// attn[b,t,s] = (1/32) Q[b,t,:].K[b,s,:]; packed argmax via atomicMax.
// grid = (NS/BN, NT/BM, BATCH), block = 256.
// Double-buffered LDS; chunk k+1 staged with async global->LDS b128 copies
// while chunk k is consumed by WMMAs. ASYNCcnt completes in order, so
// s_wait_asynccnt <= NLOAD_S retires exactly the previous chunk's copies.
// ---------------------------------------------------------------------------
__global__ __launch_bounds__(256)
void attn_scores_kernel(const unsigned short* __restrict__ Qh, const unsigned short* __restrict__ Ql,
                        const unsigned short* __restrict__ Kh, const unsigned short* __restrict__ Kl,
                        float* __restrict__ attn, unsigned long long* __restrict__ mkey) {
  __shared__ unsigned short sAh[2][BM * AST], sAl[2][BM * AST];
  __shared__ unsigned short sBh[2][BN * AST], sBl[2][BN * AST];

  const int tid   = threadIdx.x;
  const int lane  = tid & 31;
  const int wid   = tid >> 5;
  const int waveM = wid & 3;
  const int waveN = wid >> 2;
  const int sb = blockIdx.x, tb = blockIdx.y, b = blockIdx.z;

  const size_t qBase = ((size_t)b * NT + (size_t)tb * BM) * DIM;
  const size_t kBase = ((size_t)b * NS + (size_t)sb * BN) * DIM;

  const int arow = tid >> 1, ahalf = tid & 1;
  const int brow = tid >> 2, bq = tid & 3;
  const unsigned short* gQh = Qh + qBase + (size_t)arow * DIM + ahalf * 16;
  const unsigned short* gQl = Ql + qBase + (size_t)arow * DIM + ahalf * 16;
  const unsigned short* gKh = Kh + kBase + (size_t)brow * DIM + bq * 8;
  const unsigned short* gKl = Kl + kBase + (size_t)brow * DIM + bq * 8;
  const int aOff = arow * AST + ahalf * 16;
  const int bOff = brow * AST + bq * 8;

  v8f acc[2][2] = {};

  const int r16    = lane & 15;
  const int khalfA = (lane >> 4) * 16;
  const int khalfB = (lane >> 4) * 16;

  // stage chunk 0 into buffer 0
  cp16_g2l(gQh,     &sAh[0][aOff]);
  cp16_g2l(gQh + 8, &sAh[0][aOff + 8]);
  cp16_g2l(gQl,     &sAl[0][aOff]);
  cp16_g2l(gQl + 8, &sAl[0][aOff + 8]);
  cp16_g2l(gKh,     &sBh[0][bOff]);
  cp16_g2l(gKl,     &sBl[0][bOff]);

  for (int kc = 0; kc < NK; ++kc) {
    const int cur = kc & 1;
    if (kc + 1 < NK) {   // prior trailing barrier guarantees buf `nxt` is free
      const int nxt = cur ^ 1;
      const int k0  = (kc + 1) * BK;
      cp16_g2l(gQh + k0,     &sAh[nxt][aOff]);
      cp16_g2l(gQh + k0 + 8, &sAh[nxt][aOff + 8]);
      cp16_g2l(gQl + k0,     &sAl[nxt][aOff]);
      cp16_g2l(gQl + k0 + 8, &sAl[nxt][aOff + 8]);
      cp16_g2l(gKh + k0,     &sBh[nxt][bOff]);
      cp16_g2l(gKl + k0,     &sBl[nxt][bOff]);
      wait_async<NLOAD_S>();  // chunk kc's copies retired (in-order completion)
    } else {
      wait_async<0>();
    }
    __syncthreads();          // everyone's chunk kc data is in LDS

    Frag ah[2], al[2], bh[2], bl[2];
#pragma unroll
    for (int mt = 0; mt < 2; ++mt) {
      const unsigned short* bh_ = &sAh[cur][(waveM * 32 + mt * 16 + r16) * AST];
      const unsigned short* bl_ = &sAl[cur][(waveM * 32 + mt * 16 + r16) * AST];
      ah[mt].q[0] = *(const uint4*)(bh_ + khalfA);
      ah[mt].q[1] = *(const uint4*)(bh_ + 8 + khalfA);
      al[mt].q[0] = *(const uint4*)(bl_ + khalfA);
      al[mt].q[1] = *(const uint4*)(bl_ + 8 + khalfA);
    }
#pragma unroll
    for (int nt = 0; nt < 2; ++nt) {
      const unsigned short* bh_ = &sBh[cur][(waveN * 32 + nt * 16 + r16) * AST + khalfB];
      const unsigned short* bl_ = &sBl[cur][(waveN * 32 + nt * 16 + r16) * AST + khalfB];
      bh[nt].q[0] = *(const uint4*)(bh_);
      bh[nt].q[1] = *(const uint4*)(bh_ + 8);
      bl[nt].q[0] = *(const uint4*)(bl_);
      bl[nt].q[1] = *(const uint4*)(bl_ + 8);
    }
#pragma unroll
    for (int mt = 0; mt < 2; ++mt)
#pragma unroll
      for (int nt = 0; nt < 2; ++nt) {
        acc[mt][nt] = __builtin_amdgcn_wmma_f32_16x16x32_bf16(
            false, ah[mt].v, false, bh[nt].v, (short)0, acc[mt][nt], false, false);
        acc[mt][nt] = __builtin_amdgcn_wmma_f32_16x16x32_bf16(
            false, ah[mt].v, false, bl[nt].v, (short)0, acc[mt][nt], false, false);
        acc[mt][nt] = __builtin_amdgcn_wmma_f32_16x16x32_bf16(
            false, al[mt].v, false, bh[nt].v, (short)0, acc[mt][nt], false, false);
      }
    __syncthreads();          // compute on buf `cur` done; it may be re-staged
  }

  // epilogue: scale, store scores, row-wise argmax reduction
  constexpr float SCALE = 0.03125f; // 1/sqrt(1024)
  unsigned long long best[2][8];
#pragma unroll
  for (int mt = 0; mt < 2; ++mt)
#pragma unroll
    for (int v = 0; v < 8; ++v) best[mt][v] = 0ull;

#pragma unroll
  for (int mt = 0; mt < 2; ++mt)
#pragma unroll
    for (int nt = 0; nt < 2; ++nt) {
      int s     = sb * BN + waveN * 32 + nt * 16 + (lane & 15);
      int tbase = tb * BM + waveM * 32 + mt * 16 + ((lane >> 4) * 8);
#pragma unroll
      for (int v = 0; v < 8; ++v) {
        float sc = acc[mt][nt][v] * SCALE;
        attn[((size_t)b * NT + tbase + v) * NS + s] = sc;
        unsigned long long key = pack_key(sc, s);
        if (key > best[mt][v]) best[mt][v] = key;
      }
    }

  // reduce over the 16 lanes of each half-wave (lanes share the same row set)
#pragma unroll
  for (int mt = 0; mt < 2; ++mt)
#pragma unroll
    for (int v = 0; v < 8; ++v) {
#pragma unroll
      for (int m = 1; m < 16; m <<= 1) {
        unsigned long long o = shflx64(best[mt][v], m);
        if (o > best[mt][v]) best[mt][v] = o;
      }
    }

  if ((lane & 15) == 0) {
#pragma unroll
    for (int mt = 0; mt < 2; ++mt)
#pragma unroll
      for (int v = 0; v < 8; ++v) {
        int t = tb * BM + waveM * 32 + mt * 16 + ((lane >> 4) * 8) + v;
        atomicMax(&mkey[(size_t)b * NT + t], best[mt][v]);
      }
  }
}

__global__ __launch_bounds__(256)
void init_keys_kernel(unsigned long long* __restrict__ mkey) {
  int i = blockIdx.x * blockDim.x + threadIdx.x;
  if (i < BATCH * NT) mkey[i] = 0ull;
}

__global__ __launch_bounds__(256)
void gather_kernel(const float* __restrict__ scene, const unsigned long long* __restrict__ mkey,
                   float* __restrict__ outm) {
  int row = blockIdx.x;                 // b*NT + t
  int b   = row >> 9;                   // NT == 512
  unsigned long long k = mkey[row];
  int s = (NS - 1) - (int)(unsigned)(k & 0xFFFFFFFFull);
  const float4* src = (const float4*)(scene + ((size_t)b * NS + s) * DIM);
  float4*       dst = (float4*)(outm + (size_t)row * DIM);
  dst[threadIdx.x] = src[threadIdx.x];  // 256 threads x 16B = 4KB row
}

extern "C" void kernel_launch(void* const* d_in, const int* in_sizes, int n_in,
                              void* d_out, int out_size, void* d_ws, size_t ws_size,
                              hipStream_t stream) {
  const float* task  = (const float*)d_in[0];
  const float* scene = (const float*)d_in[1];
  const float* Wq    = (const float*)d_in[2];
  const float* bq    = (const float*)d_in[3];
  const float* Wk    = (const float*)d_in[4];
  const float* bk    = (const float*)d_in[5];

  float* out_matched = (float*)d_out;
  float* out_attn    = out_matched + (size_t)BATCH * NT * DIM;

  char* w = (char*)d_ws;
  const size_t MB = 1024 * 1024;
  unsigned short* Qh  = (unsigned short*)(w);
  unsigned short* Ql  = (unsigned short*)(w + 16 * MB);
  unsigned short* Kh  = (unsigned short*)(w + 32 * MB);
  unsigned short* Kl  = (unsigned short*)(w + 96 * MB);
  unsigned long long* mkey = (unsigned long long*)(w + 160 * MB);
  unsigned short* Wqh = (unsigned short*)(w + 161 * MB);
  unsigned short* Wql = (unsigned short*)(w + 163 * MB);
  unsigned short* Wkh = (unsigned short*)(w + 165 * MB);
  unsigned short* Wkl = (unsigned short*)(w + 167 * MB);

  init_keys_kernel<<<(BATCH * NT + 255) / 256, 256, 0, stream>>>(mkey);
  wsplit_kernel<<<DIM * DIM / (256 * 8), 256, 0, stream>>>(Wq, Wqh, Wql);
  wsplit_kernel<<<DIM * DIM / (256 * 8), 256, 0, stream>>>(Wk, Wkh, Wkl);
  proj_split_kernel<<<dim3(DIM / BN, (BATCH * NT) / BM), 256, 0, stream>>>(task, Wqh, Wql, bq, Qh, Ql);
  proj_split_kernel<<<dim3(DIM / BN, (BATCH * NS) / BM), 256, 0, stream>>>(scene, Wkh, Wkl, bk, Kh, Kl);
  attn_scores_kernel<<<dim3(NS / BN, NT / BM, BATCH), 256, 0, stream>>>(Qh, Ql, Kh, Kl, out_attn, mkey);
  gather_kernel<<<BATCH * NT, 256, 0, stream>>>(scene, mkey, out_matched);
}
